// DiTBlock_2181843387046
// MI455X (gfx1250) — matrix operand order repaired and checked by reading, not compiled, not run on hardware
//
#include <hip/hip_runtime.h>

typedef __bf16 bf16;
typedef __attribute__((ext_vector_type(16))) __bf16 v16bf;
typedef __attribute__((ext_vector_type(8)))  float  v8f;

#define DEV __device__ __forceinline__

// ---- WMMA fragment helpers (bf16 16x16x32, wave32 layouts per CDNA5 ISA) ----
// A (16xK=32, 16-bit): lane<16 holds row=lane, K {0..7,16..23}; lane>=16 same row,
// K {8..15,24..31}.  => two 16B loads at +kb and +kb+16 elements (kb = (lane>>4)*8).
DEV v16bf load_a_frag(const bf16* p) {
  union { v16bf v; uint4 u[2]; } r;
  r.u[0] = *(const uint4*)(p);
  r.u[1] = *(const uint4*)(p + 16);
  return r.v;
}
// B (K=32 x 16, 16-bit): lane<16 = col lane, K {0..7,16..23}; lane>=16 = same col, K {8..15,24..31}.
// => two 16B loads at +hi*8 and +hi*8+16 elements.
DEV v16bf load_b_frag(const bf16* p) {
  union { v16bf v; uint4 u[2]; } r;
  r.u[0] = *(const uint4*)(p);
  r.u[1] = *(const uint4*)(p + 16);
  return r.v;
}
DEV v8f wmma_bf16(v16bf a, v16bf b, v8f c) {
  return __builtin_amdgcn_wmma_f32_16x16x32_bf16(false, a, false, b, (short)0, c, false, false);
}

// ---- CDNA5 async copy: global -> LDS, 16B per lane, tracked by ASYNCcnt ----
// GVS mode: mem_addr = SGPR64 + VGPR32_off + inst_offset; VDST = LDS byte addr.
DEV void async_load_b128(unsigned lds_off, const void* gbase, unsigned byte_off) {
  asm volatile("global_load_async_to_lds_b128 %0, %1, %2 offset:0"
               :: "v"(lds_off), "v"(byte_off), "s"(gbase)
               : "memory");
}
DEV void wait_async_le1() { asm volatile("s_wait_asynccnt 0x1" ::: "memory"); }
DEV void wait_async_0()   { asm volatile("s_wait_asynccnt 0x0" ::: "memory"); }
DEV unsigned lds_addr_of(const void* p) { return (unsigned)(uintptr_t)p; }

// ---------------- fp32 -> bf16 weight conversion ----------------
__global__ void cvt_f32_bf16(const float* __restrict__ in, bf16* __restrict__ out, size_t n) {
  size_t i = (size_t)blockIdx.x * 256 + threadIdx.x;
  size_t stride = (size_t)gridDim.x * 256;
  for (; i < n; i += stride) out[i] = (bf16)in[i];
}

// ---------------- adaLN: silu(c) @ Wada^T + bada -> (B, 6144) fp32 ----------------
__global__ void ada_kernel(const float* __restrict__ c, const float* __restrict__ Wada,
                           const float* __restrict__ bada, float* __restrict__ ada) {
  __shared__ __align__(16) float sc[1024];
  int b  = blockIdx.x / 24;
  int nb = blockIdx.x % 24;
  int tid = threadIdx.x;
  for (int i = tid; i < 1024; i += 256) {
    float v = c[b * 1024 + i];
    sc[i] = v / (1.f + __expf(-v));              // silu
  }
  __syncthreads();
  int n = nb * 256 + tid;
  const float* wr = Wada + (size_t)n * 1024;
  float acc = 0.f;
  for (int k = 0; k < 1024; k += 4) {
    float4 w = *(const float4*)(wr + k);
    float4 s = *(const float4*)(sc + k);
    acc += w.x * s.x + w.y * s.y + w.z * s.z + w.w * s.w;
  }
  ada[(size_t)b * 6144 + n] = acc + bada[n];
}

// ---------------- rmsnorm(x)*sqrt(D)*g, then modulate -> bf16 ----------------
__global__ void normmod_kernel(const float* __restrict__ x, const float* __restrict__ g,
                               const float* __restrict__ ada, int shiftOff, int scaleOff,
                               bf16* __restrict__ out) {
  int row = blockIdx.x;            // b*T + t   (T = 1024)
  int b   = row >> 10;
  int tid = threadIdx.x;
  const float* xr = x + (size_t)row * 1024;
  float v[4];
  float ssq = 0.f;
#pragma unroll
  for (int i = 0; i < 4; i++) { v[i] = xr[tid + i * 256]; ssq += v[i] * v[i]; }
  for (int off = 16; off > 0; off >>= 1) ssq += __shfl_xor(ssq, off, 32);
  __shared__ float tmp[8];
  int lane = tid & 31, w = tid >> 5;
  if (lane == 0) tmp[w] = ssq;
  __syncthreads();
  if (tid == 0) { float s = 0.f; for (int i = 0; i < 8; i++) s += tmp[i]; tmp[0] = s; }
  __syncthreads();
  float nrm = fmaxf(sqrtf(tmp[0]), 1e-12f);
  float inv = 32.0f * g[0] / nrm;              // sqrt(1024) = 32
  const float* sh = ada + (size_t)b * 6144 + shiftOff;
  const float* sl = ada + (size_t)b * 6144 + scaleOff;
  bf16* orow = out + (size_t)row * 1024;
#pragma unroll
  for (int i = 0; i < 4; i++) {
    int d = tid + i * 256;
    orow[d] = (bf16)(v[i] * inv * (1.f + sl[d]) + sh[d]);
  }
}

// ---------------- generic bf16 GEMM:  C = A(MxK) @ W(NxK)^T + bias ----------------
// W tile (64 x 32) staged in LDS via async-to-LDS, double buffered; 8 waves share it.
// MODE 0: bias -> bf16 out.  MODE 1: gelu(tanh)(bias) -> bf16.  MODE 2: fp32 out = xres + gate*(acc+bias)
template <int MODE>
__global__ void __launch_bounds__(256) gemm_kernel(const bf16* __restrict__ A, const bf16* __restrict__ W,
                                                   const float* __restrict__ bias, void* __restrict__ outp,
                                                   const float* __restrict__ xres, const float* __restrict__ ada,
                                                   int chunkOff, int M, int N, int K) {
  __shared__ __align__(16) bf16 Wl[2][64][40];   // 80B row pitch: 16B aligned, bank-conflict free

  int tid = threadIdx.x;
  int w = tid >> 5, lane = tid & 31;
  int row = lane & 15, hi = lane >> 4, kb = hi * 8;
  int m0 = blockIdx.y * 128 + w * 16;
  int n0 = blockIdx.x * 64;
  const bf16* arow = A + (size_t)(m0 + row) * K;

  // staging map: thread -> (row of W tile, 8-elem chunk)
  int srow   = tid >> 2;          // 0..63
  int schunk = (tid & 3) * 8;     // elem 0,8,16,24
  unsigned gbase = ((unsigned)(n0 + srow) * (unsigned)K + (unsigned)schunk) * 2u;
  unsigned lds0  = lds_addr_of(&Wl[0][srow][schunk]);
  unsigned lds1  = lds_addr_of(&Wl[1][srow][schunk]);

  async_load_b128(lds0, W, gbase);               // fill buffer 0 (k0 = 0)

  v8f acc[4] = {};
  int nsteps = K >> 5;
  for (int s = 0; s < nsteps; s++) {
    int k0 = s << 5;
    bool hasNext = (s + 1 < nsteps);
    if (hasNext) {
      async_load_b128((s & 1) ? lds0 : lds1, W, gbase + (unsigned)(k0 + 32) * 2u);
      wait_async_le1();                          // current buffer's fill complete
    } else {
      wait_async_0();
    }
    __syncthreads();                             // all waves' fills visible
    const bf16 (*Wb)[40] = Wl[s & 1];
    v16bf a = load_a_frag(arow + k0 + kb);
#pragma unroll
    for (int t = 0; t < 4; t++) {
      v16bf bfr = load_b_frag(&Wb[t * 16 + row][hi * 8]);
      acc[t] = wmma_bf16(a, bfr, acc[t]);
    }
    __syncthreads();                             // done reading before next refill
  }

#pragma unroll
  for (int t = 0; t < 4; t++) {
    int n = n0 + t * 16 + row;
    float bval = bias[n];
#pragma unroll
    for (int r = 0; r < 8; r++) {
      int m = m0 + hi * 8 + r;
      float v = acc[t][r] + bval;
      size_t idx = (size_t)m * N + n;
      if (MODE == 0) {
        ((bf16*)outp)[idx] = (bf16)v;
      } else if (MODE == 1) {
        float gg = 0.5f * v * (1.f + tanhf(0.7978845608f * (v + 0.044715f * v * v * v)));
        ((bf16*)outp)[idx] = (bf16)gg;
      } else {
        int bidx = m >> 10;                     // T = 1024
        float gate = ada[(size_t)bidx * 6144 + chunkOff + n];
        ((float*)outp)[idx] = xres[idx] + gate * v;
      }
    }
  }
}

// ---------------- in-place q/k RMSNorm over head dim 64 ----------------
// q <- q/||q|| * gq   (sqrt(hd)*hd^-0.5 cancels) ;  k <- k/||k|| * 8*gk
__global__ void qknorm_kernel(bf16* __restrict__ qkv, const float* __restrict__ gq,
                              const float* __restrict__ gk) {
  int idx = blockIdx.x;                 // b*16384 + h*1024 + t
  int t = idx & 1023, h = (idx >> 10) & 15, b = idx >> 14;
  int d = threadIdx.x;                  // 64 threads
  bf16* qp = qkv + (size_t)(b * 1024 + t) * 3072 + h * 64;
  bf16* kp = qp + 1024;
  float qv = (float)qp[d], kv = (float)kp[d];
  __shared__ float red[64];
  red[d] = qv * qv; __syncthreads();
  for (int s = 32; s > 0; s >>= 1) { if (d < s) red[d] += red[d + s]; __syncthreads(); }
  float nq = fmaxf(sqrtf(red[0]), 1e-12f);
  __syncthreads();
  red[d] = kv * kv; __syncthreads();
  for (int s = 32; s > 0; s >>= 1) { if (d < s) red[d] += red[d + s]; __syncthreads(); }
  float nk = fmaxf(sqrtf(red[0]), 1e-12f);
  qp[d] = (bf16)(qv * (gq[0] / nq));
  kp[d] = (bf16)(kv * (8.0f * gk[0] / nk));
}

// ---------------- flash attention: block = (b, h, 128 q rows), 8 waves x 16-row tiles ----------------
__global__ void __launch_bounds__(256) attn_kernel(const bf16* __restrict__ qkv, bf16* __restrict__ obuf) {
  __shared__ __align__(16) bf16 Kl[64][72];        // K rows (j, d) -- filled by async-to-LDS
  __shared__ __align__(16) bf16 Vt[64][72];        // V transposed (d, j)
  __shared__ __align__(16) bf16 Pl[8][16][72];     // per-wave P staging (i, j)

  int tid = threadIdx.x;
  int w = tid >> 5, lane = tid & 31;
  int row = lane & 15, hi = lane >> 4, kb = hi * 8;
  int qt = blockIdx.x & 7;
  int h  = (blockIdx.x >> 3) & 15;
  int b  = blockIdx.x >> 7;

  int qbase = qt * 128 + w * 16;
  const bf16* qp = qkv + (size_t)(b * 1024 + qbase + row) * 3072 + h * 64;
  v16bf qf0 = load_a_frag(qp + kb);        // d = 0..31
  v16bf qf1 = load_a_frag(qp + 32 + kb);   // d = 32..63

  v8f o[4] = {};
  float mrow[8], lrow[8];
#pragma unroll
  for (int r = 0; r < 8; r++) { mrow[r] = -1e30f; lrow[r] = 0.f; }

  int jld = tid >> 2;            // 0..63 : kv row this thread loads
  int dld = (tid & 3) * 16;      // 0/16/32/48
  unsigned klds = lds_addr_of(&Kl[jld][dld]);

  for (int c = 0; c < 16; c++) {
    int keybase = c * 64;
    unsigned kgoff = (unsigned)(((unsigned)(b * 1024 + keybase + jld) * 3072u) + 1024u +
                                (unsigned)(h * 64 + dld)) * 2u;
    // K tile: async global -> LDS (no VGPR roundtrip)
    async_load_b128(klds,      qkv, kgoff);
    async_load_b128(klds + 16, qkv, kgoff + 16);
    // V tile: through registers (needs transpose into Vt)
    const bf16* vg = qkv + ((size_t)(b * 1024 + keybase + jld) * 3072) + 2048 + h * 64 + dld;
    union { uint4 u[2]; bf16 e[16]; } vv;
    vv.u[0] = *(const uint4*)(vg);
    vv.u[1] = *(const uint4*)(vg + 8);
#pragma unroll
    for (int i = 0; i < 16; i++) Vt[dld + i][jld] = vv.e[i];
    wait_async_0();
    __syncthreads();

    // S = Q K^T  (16 x 64), four 16x16 tiles
    v8f s[4] = {};
#pragma unroll
    for (int t = 0; t < 4; t++) {
      v16bf b0 = load_b_frag(&Kl[t * 16 + row][hi * 8]);
      s[t] = wmma_bf16(qf0, b0, s[t]);
      v16bf b1 = load_b_frag(&Kl[t * 16 + row][32 + hi * 8]);
      s[t] = wmma_bf16(qf1, b1, s[t]);
    }

    // streaming softmax (row stats live in each 16-lane half of the wave)
#pragma unroll
    for (int r = 0; r < 8; r++) {
      float vmax = fmaxf(fmaxf(s[0][r], s[1][r]), fmaxf(s[2][r], s[3][r]));
#pragma unroll
      for (int off = 1; off < 16; off <<= 1) vmax = fmaxf(vmax, __shfl_xor(vmax, off, 32));
      float mnew = fmaxf(mrow[r], vmax);
      float corr = __expf(mrow[r] - mnew);
      mrow[r] = mnew;
      float rs = 0.f;
#pragma unroll
      for (int t = 0; t < 4; t++) { float p = __expf(s[t][r] - mnew); s[t][r] = p; rs += p; }
#pragma unroll
      for (int off = 1; off < 16; off <<= 1) rs += __shfl_xor(rs, off, 32);
      lrow[r] = lrow[r] * corr + rs;
#pragma unroll
      for (int t = 0; t < 4; t++) o[t][r] *= corr;
    }

    // P: C-layout -> A-layout via per-wave LDS
#pragma unroll
    for (int t = 0; t < 4; t++)
#pragma unroll
      for (int r = 0; r < 8; r++)
        Pl[w][hi * 8 + r][t * 16 + row] = (bf16)s[t][r];

    v16bf pf0 = load_a_frag(&Pl[w][row][kb]);
    v16bf pf1 = load_a_frag(&Pl[w][row][32 + kb]);
#pragma unroll
    for (int t2 = 0; t2 < 4; t2++) {
      v16bf vf0 = load_b_frag(&Vt[t2 * 16 + row][hi * 8]);
      o[t2] = wmma_bf16(pf0, vf0, o[t2]);
      v16bf vf1 = load_b_frag(&Vt[t2 * 16 + row][32 + hi * 8]);
      o[t2] = wmma_bf16(pf1, vf1, o[t2]);
    }
    __syncthreads();
  }

#pragma unroll
  for (int r = 0; r < 8; r++) {
    float inv = 1.f / lrow[r];
    int qtok = qbase + hi * 8 + r;
    bf16* orow = obuf + (size_t)(b * 1024 + qtok) * 1024 + h * 64;
#pragma unroll
    for (int t2 = 0; t2 < 4; t2++)
      orow[t2 * 16 + row] = (bf16)(o[t2][r] * inv);
  }
}

extern "C" void kernel_launch(void* const* d_in, const int* in_sizes, int n_in,
                              void* d_out, int out_size, void* d_ws, size_t ws_size,
                              hipStream_t stream) {
  (void)in_sizes; (void)n_in; (void)out_size; (void)ws_size;
  const float* x     = (const float*)d_in[0];
  const float* c     = (const float*)d_in[1];
  const float* g1    = (const float*)d_in[2];
  const float* g2    = (const float*)d_in[3];
  const float* gq    = (const float*)d_in[4];
  const float* gk    = (const float*)d_in[5];
  const float* Wqkv  = (const float*)d_in[6];
  const float* bqkv  = (const float*)d_in[7];
  const float* Wproj = (const float*)d_in[8];
  const float* bproj = (const float*)d_in[9];
  const float* Wfc1  = (const float*)d_in[10];
  const float* bfc1  = (const float*)d_in[11];
  const float* Wfc2  = (const float*)d_in[12];
  const float* bfc2  = (const float*)d_in[13];
  const float* Wada  = (const float*)d_in[14];
  const float* bada  = (const float*)d_in[15];

  char* ws = (char*)d_ws;
  const size_t MB = 1u << 20;
  float* ada  = (float*)ws;                                   // 192 KB
  bf16*  hbuf = (bf16*)(ws + 256 * 1024);                     // 16 MB
  bf16*  qkvb = (bf16*)(ws + 256 * 1024 + 16 * MB);           // 48 MB
  bf16*  obuf = (bf16*)(ws + 256 * 1024 + 64 * MB);           // 16 MB
  bf16*  h4   = qkvb;                                         // reuse qkv+obuf span (64 MB)
  float* xat  = (float*)(ws + 256 * 1024 + 80 * MB);          // 32 MB
  bf16*  wq   = (bf16*)(ws + 256 * 1024 + 112 * MB);          // 6 MB
  bf16*  wp   = wq + (size_t)3072 * 1024;                     // 2 MB
  bf16*  w1   = wp + (size_t)1024 * 1024;                     // 8 MB
  bf16*  w2   = w1 + (size_t)4096 * 1024;                     // 8 MB
  float* outF = (float*)d_out;

  cvt_f32_bf16<<<2048, 256, 0, stream>>>(Wqkv,  wq, (size_t)3072 * 1024);
  cvt_f32_bf16<<<1024, 256, 0, stream>>>(Wproj, wp, (size_t)1024 * 1024);
  cvt_f32_bf16<<<2048, 256, 0, stream>>>(Wfc1,  w1, (size_t)4096 * 1024);
  cvt_f32_bf16<<<2048, 256, 0, stream>>>(Wfc2,  w2, (size_t)4096 * 1024);

  ada_kernel<<<192, 256, 0, stream>>>(c, Wada, bada, ada);

  // attention branch
  normmod_kernel<<<8192, 256, 0, stream>>>(x, g1, ada, 0, 1024, hbuf);
  gemm_kernel<0><<<dim3(48, 64), 256, 0, stream>>>(hbuf, wq, bqkv, qkvb, nullptr, nullptr, 0, 8192, 3072, 1024);
  qknorm_kernel<<<131072, 64, 0, stream>>>(qkvb, gq, gk);
  attn_kernel<<<1024, 256, 0, stream>>>(qkvb, obuf);
  gemm_kernel<2><<<dim3(16, 64), 256, 0, stream>>>(obuf, wp, bproj, xat, x, ada, 2 * 1024, 8192, 1024, 1024);

  // MLP branch
  normmod_kernel<<<8192, 256, 0, stream>>>(xat, g2, ada, 3 * 1024, 4 * 1024, hbuf);
  gemm_kernel<1><<<dim3(64, 64), 256, 0, stream>>>(hbuf, w1, bfc1, h4, nullptr, nullptr, 0, 8192, 4096, 1024);
  gemm_kernel<2><<<dim3(16, 64), 256, 0, stream>>>(h4, w2, bfc2, outF, xat, ada, 5 * 1024, 8192, 1024, 4096);
}
